// Union_Loss_16655883174017
// MI455X (gfx1250) — matrix operand order repaired and checked by reading, not compiled, not run on hardware
//
#include <hip/hip_runtime.h>
#include <hip/hip_bf16.h>

typedef __attribute__((ext_vector_type(16))) _Float16 v16h;
typedef __attribute__((ext_vector_type(8)))  float    v8f;

#define B_IMGS 8
#define A_ANCH 100000
#define N_ANN  32
#define C_CLS  24
#define ANN_DIM 37
#define TILES_PER_IMG (A_ANCH / 16)     // 6250 (exact)
#define WAVES_PER_BLOCK 8
#define BLOCK_THREADS 256

#define LN2F 0.69314718055994530942f

__device__ __forceinline__ float fast_rcp(float x)  { return __builtin_amdgcn_rcpf(x); }
__device__ __forceinline__ float fast_sqrt(float x) { return __builtin_amdgcn_sqrtf(x); }
__device__ __forceinline__ float fast_log2(float x) { return __builtin_amdgcn_logf(x); }
__device__ __forceinline__ float fast_ln(float x)   { return fast_log2(x) * LN2F; }

__device__ __forceinline__ float smooth_l1(float d) {
    d = fabsf(d);
    return (d <= (1.0f / 9.0f)) ? 4.5f * d * d : d - (0.5f / 9.0f);
}

__global__ void ul_zero_ws(float* ws) {
    int i = threadIdx.x;
    if (i < 64) ws[i] = 0.0f;
}

__global__ __launch_bounds__(BLOCK_THREADS) void ul_main(
    const float* __restrict__ actCls,   // (B, A, C)
    const float* __restrict__ subReg,   // (B, A, 4)
    const float* __restrict__ objReg,   // (B, A, 4)
    const float* __restrict__ anchors,  // (1, A, 4)
    const float* __restrict__ ann,      // (B, N, 37)
    float* __restrict__ ws)             // 8 images x 8 slots
{
    __shared__ float sAnn[N_ANN * ANN_DIM];
    __shared__ float sAreaU[N_ANN];     // union-box area
    __shared__ float sInvS[N_ANN];      // 1 / max(area_sub, 1e-8)
    __shared__ float sInvO[N_ANN];      // 1 / max(area_obj, 1e-8)
    __shared__ int   sArg[WAVES_PER_BLOCK * 16];
    __shared__ float sPos[WAVES_PER_BLOCK * 16];
    __shared__ float sPosReg[WAVES_PER_BLOCK * 16];

    const int b   = blockIdx.y;
    const int tid = threadIdx.x;

    // stage this image's 32 annotations into LDS
    const float* annB = ann + (size_t)b * N_ANN * ANN_DIM;
    for (int i = tid; i < N_ANN * ANN_DIM; i += BLOCK_THREADS) sAnn[i] = annB[i];
    // per-annotation derived quantities (independent of staging -> same barrier)
    if (tid < N_ANN) {
        const float* row = annB + tid * ANN_DIM;
        float s0 = row[0], s1 = row[1], s2 = row[2], s3 = row[3];
        float o0 = row[4], o1 = row[5], o2 = row[6], o3 = row[7];
        float u0 = row[8], u1 = row[9], u2 = row[10], u3 = row[11];
        sAreaU[tid] = (u2 - u0) * (u3 - u1);
        sInvS[tid]  = fast_rcp(fmaxf((s2 - s0) * (s3 - s1), 1e-8f));
        sInvO[tid]  = fast_rcp(fmaxf((o2 - o0) * (o3 - o1), 1e-8f));
    }
    __syncthreads();

    const int wave = tid >> 5;
    const int lane = tid & 31;
    const int tile = blockIdx.x * WAVES_PER_BLOCK + wave;

    float accAct = 0.f, accNumPos = 0.f, accSub = 0.f, accObj = 0.f, accDiff = 0.f, accCnt = 0.f;

    if (tile < TILES_PER_IMG) {   // uniform per wave -> EXEC all-ones inside
        const int m    = lane & 15;        // A-matrix row (anchor within tile)
        const int half = lane >> 4;        // which K-half this lane owns
        const int aIdx = tile * 16 + m;    // anchor index

        const float a0 = anchors[aIdx * 4 + 0];
        const float a1 = anchors[aIdx * 4 + 1];
        const float a2 = anchors[aIdx * 4 + 2];
        const float a3 = anchors[aIdx * 4 + 3];
        const float awd = a3 - a1, aht = a2 - a0;
        const float acx = a1 + 0.5f * awd, acy = a0 + 0.5f * aht;
        const float invAw = fast_rcp(awd), invAh = fast_rcp(aht);
        const float aArea = aht * awd;

        // ---- per-lane: 16 annotations, build A fragment (union_iou as f16) ----
        // lanes 0-15: K in {0..7, 16..23}; lanes 16-31: K in {8..15, 24..31}
        v16h afrag;
        float bestS = -1.0f;
        int bestI = 0, anyU = 0;
        const int kb0 = half ? 8 : 0;
        const int kb1 = half ? 24 : 16;
        #pragma unroll
        for (int r = 0; r < 8; ++r) {
            const int kpair = (r < 4) ? (kb0 + 2 * r) : (kb1 + 2 * (r - 4));
            #pragma unroll
            for (int j = 0; j < 2; ++j) {
                const int kk = kpair + j;                 // ascending order scan
                const float* row = &sAnn[kk * ANN_DIM];
                // IoU vs union box ann[8:12]
                float u0 = row[8], u1 = row[9], u2 = row[10], u3 = row[11];
                float iw = fmaxf(fminf(a3, u2) - fmaxf(a1, u0), 0.f);
                float ih = fmaxf(fminf(a2, u3) - fmaxf(a0, u1), 0.f);
                float inter = iw * ih;
                float iou = inter * fast_rcp(fmaxf(aArea + sAreaU[kk] - inter, 1e-8f));
                // IoA vs sub box ann[0:4]
                float s0 = row[0], s1 = row[1], s2 = row[2], s3 = row[3];
                float iws = fmaxf(fminf(a3, s2) - fmaxf(a1, s0), 0.f);
                float ihs = fmaxf(fminf(a2, s3) - fmaxf(a0, s1), 0.f);
                float ioaS = iws * ihs * sInvS[kk];
                // IoA vs obj box ann[4:8]
                float o0 = row[4], o1 = row[5], o2 = row[6], o3 = row[7];
                float iwo = fmaxf(fminf(a3, o2) - fmaxf(a1, o0), 0.f);
                float iho = fmaxf(fminf(a2, o3) - fmaxf(a0, o1), 0.f);
                float ioaO = iwo * iho * sInvO[kk];

                int u = (iou > 0.25f) && (ioaS > 0.25f) && (ioaO > 0.25f);
                float score = u ? 0.5f * (iou + fast_sqrt(ioaS * ioaO)) : 0.f;
                anyU |= u;
                if (score > bestS) { bestS = score; bestI = kk; }  // first-occurrence argmax
                afrag[2 * r + j] = (_Float16)(u ? 1.0f : 0.0f);
            }
        }

        // ---- combine lane pair (m, m+16): full 32-annotation max/argmax/any ----
        float oS = __shfl_xor(bestS, 16, 32);
        int   oI = __shfl_xor(bestI, 16, 32);
        int   oU = __shfl_xor(anyU, 16, 32);
        float scMax; int argF;
        if (oS > bestS)      { scMax = oS;    argF = oI; }
        else if (bestS > oS) { scMax = bestS; argF = bestI; }
        else                 { scMax = bestS; argF = (bestI < oI) ? bestI : oI; }
        const int   posF    = anyU | oU;
        const float posRegF = (scMax >= 0.1f) ? 1.f : 0.f;

        if (half == 0) {
            sArg[wave * 16 + m]    = argF;
            sPos[wave * 16 + m]    = posF ? 1.f : 0.f;
            sPosReg[wave * 16 + m] = posRegF;
        }
        __builtin_amdgcn_wave_barrier();

        // ---- B fragments: labels (32 x 16 classes), two tiles covering C=24 ----
        // B layout: VGPR r (r<4), lanes 0-15 -> K=2r,2r+1; lanes 16-31 -> K=8+2r,+1; r>=4 adds 8 (col N=lane%16)
        v16h bfrag0, bfrag1;
        {
            const int c0 = m, c1 = 16 + m;
            const int kbB = half ? 8 : 0;
            #pragma unroll
            for (int r = 0; r < 8; ++r) {
                const int k0 = kbB + 2 * r + ((r < 4) ? 0 : 8), k1 = k0 + 1;
                bfrag0[2 * r]     = (_Float16)sAnn[k0 * ANN_DIM + 13 + c0];
                bfrag0[2 * r + 1] = (_Float16)sAnn[k1 * ANN_DIM + 13 + c0];
                float l0 = (c1 < C_CLS) ? sAnn[k0 * ANN_DIM + 13 + c1] : 0.f;
                float l1 = (c1 < C_CLS) ? sAnn[k1 * ANN_DIM + 13 + c1] : 0.f;
                bfrag1[2 * r]     = (_Float16)l0;
                bfrag1[2 * r + 1] = (_Float16)l1;
            }
        }

        // all_fore tiles: (16x32 u) x (32x16 labels) -> 16x16 f32, exact in f16/f32
        v8f d0 = {}; v8f d1 = {};
        d0 = __builtin_amdgcn_wmma_f32_16x16x32_f16(false, afrag, false, bfrag0,
                                                    (short)0, d0, false, false);
        d1 = __builtin_amdgcn_wmma_f32_16x16x32_f16(false, afrag, false, bfrag1,
                                                    (short)0, d1, false, false);

        // ---- focal loss: lane covers class c (column), rows r (+8 for upper half) ----
        const float* clsB = actCls + (size_t)b * A_ANCH * C_CLS;
        #pragma unroll
        for (int r = 0; r < 8; ++r) {
            const int rowm = r + (half ? 8 : 0);            // D-matrix row -> anchor
            const int aRow = tile * 16 + rowm;
            const int argR  = sArg[wave * 16 + rowm];
            const float posR = sPos[wave * 16 + rowm];
            // tile 0: class m
            {
                float af = fminf(fmaxf(d0[r], 0.f), 1.f);
                float aa = sAnn[argR * ANN_DIM + 13 + m];
                float t  = 2.f * aa - af;                   // assigned - (all_fore - assigned)
                float cv = fminf(fmaxf(clsB[(size_t)aRow * C_CLS + m], 1e-4f), 1.f - 1e-4f);
                bool one = (t == 1.f);
                float alphaF = one ? 0.25f : 0.75f;
                float pt = one ? (1.f - cv) : cv;
                float fw = alphaF * pt * pt;
                float l2c = fast_log2(cv), l2m = fast_log2(1.f - cv);
                float bce = -LN2F * (l2m + t * (l2c - l2m));
                accAct += ((t != -1.f) ? fw * bce : 0.f) * posR;
            }
            // tile 1: class 16+m (only valid for m < 8)
            const int c1 = 16 + m;
            if (c1 < C_CLS) {
                float af = fminf(fmaxf(d1[r], 0.f), 1.f);
                float aa = sAnn[argR * ANN_DIM + 13 + c1];
                float t  = 2.f * aa - af;
                float cv = fminf(fmaxf(clsB[(size_t)aRow * C_CLS + c1], 1e-4f), 1.f - 1e-4f);
                bool one = (t == 1.f);
                float alphaF = one ? 0.25f : 0.75f;
                float pt = one ? (1.f - cv) : cv;
                float fw = alphaF * pt * pt;
                float l2c = fast_log2(cv), l2m = fast_log2(1.f - cv);
                float bce = -LN2F * (l2m + t * (l2c - l2m));
                accAct += ((t != -1.f) ? fw * bce : 0.f) * posR;
            }
        }

        // ---- regression losses: one lane per anchor (half 0) ----
        if (half == 0) {
            const float* arow = &sAnn[argF * ANN_DIM];
            float s0 = arow[0], s1 = arow[1], s2 = arow[2], s3 = arow[3];
            float o0 = arow[4], o1 = arow[5], o2 = arow[6], o3 = arow[7];
            float sr[4], orr[4];
            #pragma unroll
            for (int j = 0; j < 4; ++j) {
                sr[j]  = subReg[((size_t)b * A_ANCH + aIdx) * 4 + j];
                orr[j] = objReg[((size_t)b * A_ANCH + aIdx) * 4 + j];
            }
            float ts[4], to[4];
            {   // reg_targets(sub box)
                float w = s2 - s0, h = s3 - s1;
                float cx = s0 + 0.5f * w, cy = s1 + 0.5f * h;
                float wc = fmaxf(w, 1.f), hc = fmaxf(h, 1.f);
                ts[0] = (cy - acy) * invAh; ts[1] = (cx - acx) * invAw;
                ts[2] = fast_ln(hc * invAh); ts[3] = fast_ln(wc * invAw);
            }
            {   // reg_targets(obj box)
                float w = o2 - o0, h = o3 - o1;
                float cx = o0 + 0.5f * w, cy = o1 + 0.5f * h;
                float wc = fmaxf(w, 1.f), hc = fmaxf(h, 1.f);
                to[0] = (cy - acy) * invAh; to[1] = (cx - acx) * invAw;
                to[2] = fast_ln(hc * invAh); to[3] = fast_ln(wc * invAw);
            }
            float dx = (o0 + o2) * 0.5f - (s0 + s2) * 0.5f;
            float dy = (o1 + o3) * 0.5f - (s1 + s3) * 0.5f;
            float wo = fmaxf(o2 - o0, 1.f), wsb = fmaxf(s2 - s0, 1.f);
            float ho = fmaxf(o3 - o1, 1.f), hsb = fmaxf(s3 - s1, 1.f);
            float dt[4] = { dy * invAh, dx * invAw,
                            fast_ln(ho * fast_rcp(hsb)), fast_ln(wo * fast_rcp(wsb)) };
            float subL = 0.f, objL = 0.f, diffL = 0.f;
            #pragma unroll
            for (int j = 0; j < 4; ++j) {
                subL  += smooth_l1(ts[j] - sr[j]);
                objL  += smooth_l1(to[j] - orr[j]);
                diffL += smooth_l1(dt[j] - (orr[j] - sr[j]));
            }
            accSub  += subL * posRegF;
            accObj  += objL * posRegF;
            accDiff += diffL * posRegF;
            accNumPos += posF ? 1.f : 0.f;
            accCnt    += posRegF;
        }
    }

    // ---- wave32 tree reduction of the 6 accumulators, then global atomics ----
    #pragma unroll
    for (int off = 16; off >= 1; off >>= 1) {
        accAct    += __shfl_xor(accAct, off, 32);
        accNumPos += __shfl_xor(accNumPos, off, 32);
        accSub    += __shfl_xor(accSub, off, 32);
        accObj    += __shfl_xor(accObj, off, 32);
        accDiff   += __shfl_xor(accDiff, off, 32);
        accCnt    += __shfl_xor(accCnt, off, 32);
    }
    if (lane == 0) {
        float* w = ws + b * 8;
        atomicAdd(&w[0], accAct);
        atomicAdd(&w[1], accNumPos);
        atomicAdd(&w[2], accSub);
        atomicAdd(&w[3], accObj);
        atomicAdd(&w[4], accDiff);
        atomicAdd(&w[5], accCnt);
    }
}

__global__ void ul_finalize(const float* __restrict__ ws, float* __restrict__ out) {
    float al = 0.f, sl = 0.f, ol = 0.f, dl = 0.f;
    #pragma unroll
    for (int b = 0; b < B_IMGS; ++b) {
        const float* w = ws + b * 8;
        float np  = fmaxf(w[1], 1.f);
        float cnt = fmaxf(w[5], 1.f);
        al += w[0] / np;
        sl += w[2] / (4.f * cnt);
        ol += w[3] / (4.f * cnt);
        dl += w[4] / (4.f * cnt);
    }
    const float inv = 1.0f / (float)B_IMGS;
    out[0] = al * inv; out[1] = sl * inv; out[2] = ol * inv; out[3] = dl * inv;
}

extern "C" void kernel_launch(void* const* d_in, const int* in_sizes, int n_in,
                              void* d_out, int out_size, void* d_ws, size_t ws_size,
                              hipStream_t stream) {
    const float* actCls  = (const float*)d_in[0];  // (8, 100000, 24)
    const float* subReg  = (const float*)d_in[1];  // (8, 100000, 4)
    const float* objReg  = (const float*)d_in[2];  // (8, 100000, 4)
    const float* anchors = (const float*)d_in[3];  // (1, 100000, 4)
    const float* ann     = (const float*)d_in[4];  // (8, 32, 37)
    float* ws  = (float*)d_ws;
    float* out = (float*)d_out;

    ul_zero_ws<<<1, 64, 0, stream>>>(ws);

    dim3 grid((TILES_PER_IMG + WAVES_PER_BLOCK - 1) / WAVES_PER_BLOCK, B_IMGS);
    ul_main<<<grid, BLOCK_THREADS, 0, stream>>>(actCls, subReg, objReg, anchors, ann, ws);

    ul_finalize<<<1, 1, 0, stream>>>(ws, out);
}
